// Equivariance_24154896073444
// MI455X (gfx1250) — matrix operand order repaired and checked by reading, not compiled, not run on hardware
//
#include <hip/hip_runtime.h>
#include <hip/hip_bf16.h>
#include <math.h>

// ---------------------------------------------------------------------------
// Equivariance: out[i,j,y] = sum_x phi[i,(j+T0)%Nt,(x-T0)%Nx] * W[x,y]
//               logDet[i]  = Nt * log|det W|
// Strategy: fold both rolls into the A-operand gather of a batched GEMM and
// run it through v_wmma_f32_16x16x32_bf16 with 2-term float splitting
// (hi*hi + hi*lo + lo*hi) for near-f32 accuracy at bf16-WMMA throughput.
// ---------------------------------------------------------------------------

typedef __bf16 bf16_t;
typedef __attribute__((ext_vector_type(8)))  bf16_t v8bf;
typedef __attribute__((ext_vector_type(16))) bf16_t v16bf;
typedef __attribute__((ext_vector_type(8)))  float  v8f;

#define NSAMP 256
#define NT    512
#define NX    512
#define MBLK  128
#define NBLK  128
#define KT    32
#define LDSS  40   // bf16 elems per LDS row: 32 + 8 pad (keeps 16B alignment, breaks bank cycle)

__device__ __forceinline__ void split2bf16(float x, bf16_t& hi, bf16_t& lo) {
  unsigned u = __float_as_uint(x);
  unsigned short h = (unsigned short)(u >> 16);              // truncate to bf16
  float fh = __uint_as_float(((unsigned)h) << 16);
  float r  = x - fh;                                         // exact residual
  unsigned short l = (unsigned short)(__float_as_uint(r) >> 16);
  hi = __builtin_bit_cast(bf16_t, h);
  lo = __builtin_bit_cast(bf16_t, l);
}

// --------------------------- T0 per sample ---------------------------------
__global__ __launch_bounds__(32)
void t0_kernel(const float* __restrict__ phi, int* __restrict__ T0) {
  const int i    = blockIdx.x;
  const int lane = threadIdx.x;
  const float* p = phi + (size_t)i * NT * NX;
  float bv0 = INFINITY, bv1 = INFINITY;
  int   bi0 = 0x7fffffff, bi1 = 0x7fffffff;
  for (int t = lane; t < NT; t += 32) {
    float v0 = fabsf(p[(size_t)t * NX + 0]);
    float v1 = fabsf(p[(size_t)t * NX + 1]);
    if (v0 < bv0) { bv0 = v0; bi0 = t; }
    if (v1 < bv1) { bv1 = v1; bi1 = t; }
  }
  for (int off = 16; off > 0; off >>= 1) {
    float ov0 = __shfl_down(bv0, off, 32);
    int   oi0 = __shfl_down(bi0, off, 32);
    float ov1 = __shfl_down(bv1, off, 32);
    int   oi1 = __shfl_down(bi1, off, 32);
    if (ov0 < bv0 || (ov0 == bv0 && oi0 < bi0)) { bv0 = ov0; bi0 = oi0; }
    if (ov1 < bv1 || (ov1 == bv1 && oi1 < bi1)) { bv1 = ov1; bi1 = oi1; }
  }
  if (lane == 0) T0[i] = (bi0 < bi1) ? bi0 : bi1;   // min(t[:,0], t[:,1])
}

// ------------------- slogdet(W) via single-WG pivoted LU --------------------
__global__ __launch_bounds__(1024)
void logdet_kernel(const float* __restrict__ W, float* __restrict__ A,
                   float* __restrict__ logDetOut) {
  const int n   = NX;
  const int tid = threadIdx.x;
  const int nth = blockDim.x;
  __shared__ float s_red[1024];
  __shared__ int   s_idx[1024];
  __shared__ float s_logsum;
  for (int e = tid; e < n * n; e += nth) A[e] = W[e];
  if (tid == 0) s_logsum = 0.f;
  __syncthreads();
  for (int k = 0; k < n; ++k) {
    // partial pivot search over column k, rows k..n-1
    float bv = -1.f; int bi = k;
    for (int r = k + tid; r < n; r += nth) {
      float v = fabsf(A[(size_t)r * n + k]);
      if (v > bv) { bv = v; bi = r; }
    }
    s_red[tid] = bv; s_idx[tid] = bi;
    __syncthreads();
    for (int off = nth >> 1; off > 0; off >>= 1) {
      if (tid < off && s_red[tid + off] > s_red[tid]) {
        s_red[tid] = s_red[tid + off]; s_idx[tid] = s_idx[tid + off];
      }
      __syncthreads();
    }
    const int piv = s_idx[0];
    if (piv != k) {
      for (int c = tid; c < n; c += nth) {
        float t = A[(size_t)k * n + c];
        A[(size_t)k * n + c]   = A[(size_t)piv * n + c];
        A[(size_t)piv * n + c] = t;
      }
    }
    __syncthreads();
    const float akk = A[(size_t)k * n + k];
    if (tid == 0) s_logsum += logf(fabsf(akk));
    const float inv = 1.0f / akk;
    for (int r = k + 1 + tid; r < n; r += nth) {
      float f = A[(size_t)r * n + k] * inv;
      for (int c = k + 1; c < n; ++c)
        A[(size_t)r * n + c] -= f * A[(size_t)k * n + c];
    }
    __syncthreads();
  }
  const float ld = (float)NT * s_logsum;
  for (int e = tid; e < NSAMP; e += nth) logDetOut[e] = ld;
}

// ----------------------- fused roll + GEMM (WMMA) ---------------------------
__global__ __launch_bounds__(256)
void equiv_gemm_kernel(const float* __restrict__ phi,
                       const float* __restrict__ W,
                       const int*   __restrict__ T0,
                       float*       __restrict__ out) {
  __shared__ __attribute__((aligned(16))) bf16_t sA_hi[MBLK * LDSS];
  __shared__ __attribute__((aligned(16))) bf16_t sA_lo[MBLK * LDSS];
  __shared__ __attribute__((aligned(16))) bf16_t sB_hi[NBLK * LDSS];
  __shared__ __attribute__((aligned(16))) bf16_t sB_lo[NBLK * LDSS];

  const int n0g = blockIdx.x * NBLK;
  const int m0g = blockIdx.y * MBLK;
  const int i   = blockIdx.z;
  const int t0  = T0[i];

  const int tid  = threadIdx.x;
  const int lane = tid & 31;
  const int wave = tid >> 5;
  const int wm   = wave & 1;    // 2 waves along M -> 64-row slab each
  const int wn   = wave >> 1;   // 4 waves along N -> 32-col slab each
  const int half = lane >> 4;
  const int l15  = lane & 15;

  const float* phi_i = phi + (size_t)i * NT * NX;

  v8f acc[4][2];
#pragma unroll
  for (int mt = 0; mt < 4; ++mt)
#pragma unroll
    for (int nt = 0; nt < 2; ++nt)
      acc[mt][nt] = (v8f){0.f, 0.f, 0.f, 0.f, 0.f, 0.f, 0.f, 0.f};

  for (int kg = 0; kg < NX; kg += KT) {
    __syncthreads();
    // Stage A tile: A[m][k] = phi[i, (m0g+m+T0)%Nt, (kg+k-T0)%Nx], split hi/lo
    for (int e = tid; e < MBLK * KT; e += 256) {
      int m = e >> 5;
      int k = e & (KT - 1);
      int row = (m0g + m + t0) & (NT - 1);
      int col = (kg + k - t0 + NX) & (NX - 1);
      float v = phi_i[(size_t)row * NX + col];
      bf16_t h, l; split2bf16(v, h, l);
      sA_hi[m * LDSS + k] = h;
      sA_lo[m * LDSS + k] = l;
    }
    // Stage W tile transposed: sB[n][k] = W[kg+k][n0g+n] (contiguous K per lane)
    for (int e = tid; e < NBLK * KT; e += 256) {
      int n = e & (NBLK - 1);
      int k = e >> 7;
      float v = W[(size_t)(kg + k) * NX + (n0g + n)];
      bf16_t h, l; split2bf16(v, h, l);
      sB_hi[n * LDSS + k] = h;
      sB_lo[n * LDSS + k] = l;
    }
    __syncthreads();

    // B fragments: lane l holds N = l&15, K = half*16 .. +15 (contiguous)
    v16bf bh[2], bl[2];
#pragma unroll
    for (int nt = 0; nt < 2; ++nt) {
      int ncol = wn * 32 + nt * 16 + l15;
      const bf16_t* bp = &sB_hi[ncol * LDSS + half * 16];
      v8bf b0 = *(const v8bf*)(bp);
      v8bf b1 = *(const v8bf*)(bp + 8);
      bh[nt] = __builtin_shufflevector(b0, b1, 0,1,2,3,4,5,6,7,8,9,10,11,12,13,14,15);
      const bf16_t* bq = &sB_lo[ncol * LDSS + half * 16];
      v8bf c0 = *(const v8bf*)(bq);
      v8bf c1 = *(const v8bf*)(bq + 8);
      bl[nt] = __builtin_shufflevector(c0, c1, 0,1,2,3,4,5,6,7,8,9,10,11,12,13,14,15);
    }

#pragma unroll
    for (int mt = 0; mt < 4; ++mt) {
      // A fragment: lane l holds M = l&15, K = half*16 .. +15 (contiguous)
      int mrow = wm * 64 + mt * 16 + l15;
      const bf16_t* ap = &sA_hi[mrow * LDSS];
      v8bf a0 = *(const v8bf*)(ap + half * 16);
      v8bf a1 = *(const v8bf*)(ap + half * 16 + 8);
      v16bf ah = __builtin_shufflevector(a0, a1, 0,1,2,3,4,5,6,7,8,9,10,11,12,13,14,15);
      const bf16_t* aq = &sA_lo[mrow * LDSS];
      v8bf a2 = *(const v8bf*)(aq + half * 16);
      v8bf a3 = *(const v8bf*)(aq + half * 16 + 8);
      v16bf al = __builtin_shufflevector(a2, a3, 0,1,2,3,4,5,6,7,8,9,10,11,12,13,14,15);
#pragma unroll
      for (int nt = 0; nt < 2; ++nt) {
        acc[mt][nt] = __builtin_amdgcn_wmma_f32_16x16x32_bf16(
            false, ah, false, bh[nt], (short)0, acc[mt][nt], false, false);
        acc[mt][nt] = __builtin_amdgcn_wmma_f32_16x16x32_bf16(
            false, ah, false, bl[nt], (short)0, acc[mt][nt], false, false);
        acc[mt][nt] = __builtin_amdgcn_wmma_f32_16x16x32_bf16(
            false, al, false, bh[nt], (short)0, acc[mt][nt], false, false);
      }
    }
  }

  // Epilogue: C layout — VGPR r: M = r + 8*half, N = lane&15. Plain store.
#pragma unroll
  for (int mt = 0; mt < 4; ++mt) {
#pragma unroll
    for (int nt = 0; nt < 2; ++nt) {
      int j0 = m0g + wm * 64 + mt * 16 + half * 8;
      int nn = n0g + wn * 32 + nt * 16 + l15;
      float* op = out + ((size_t)i * NT + j0) * NX + nn;
#pragma unroll
      for (int r = 0; r < 8; ++r)
        op[(size_t)r * NX] = acc[mt][nt][r];
    }
  }
}

// ---------------------------------------------------------------------------
extern "C" void kernel_launch(void* const* d_in, const int* in_sizes, int n_in,
                              void* d_out, int out_size, void* d_ws, size_t ws_size,
                              hipStream_t stream) {
  const float* phi = (const float*)d_in[0];   // (256, 512, 512) f32
  const float* W   = (const float*)d_in[1];   // (512, 512) f32
  float* out    = (float*)d_out;                              // 256*512*512
  float* logDet = out + (size_t)NSAMP * NT * NX;              // + 256

  // workspace: [0,1KB) T0 ints, [4KB, 4KB+1MB) LU scratch
  int*   T0 = (int*)d_ws;
  float* LU = (float*)((char*)d_ws + 4096);

  t0_kernel<<<NSAMP, 32, 0, stream>>>(phi, T0);
  logdet_kernel<<<1, 1024, 0, stream>>>(W, LU, logDet);

  dim3 grid(NX / NBLK, NT / MBLK, NSAMP);   // 4 x 4 x 256
  equiv_gemm_kernel<<<grid, 256, 0, stream>>>(phi, W, T0, out);
}
